// QuokkaBlock_62259845923215
// MI455X (gfx1250) — compile-verified, matrix operand order checked by reading, not run on hardware
//
#include <hip/hip_runtime.h>
#include <hip/hip_bf16.h>
#include <cstdint>

#define B_   2
#define L_   1024
#define DM   1024
#define DS   32
#define DI   2048
#define DTR  64
#define HID  1408
#define XPN  128   // DT_RANK + 2*D_STATE

typedef _Float16 half8 __attribute__((ext_vector_type(8)));
typedef _Float16 v16h  __attribute__((ext_vector_type(16)));
typedef float    v8f   __attribute__((ext_vector_type(8)));
typedef unsigned int v4u __attribute__((ext_vector_type(4)));
typedef int      v8i   __attribute__((ext_vector_type(8)));
typedef int      v4i   __attribute__((ext_vector_type(4)));
typedef __attribute__((address_space(3))) _Float16 lds_f16;

#if defined(__has_builtin)
#if __has_builtin(__builtin_amdgcn_tensor_load_to_lds) && __has_builtin(__builtin_amdgcn_s_wait_tensorcnt)
#define HAVE_TDM 1
#endif
#endif

static __device__ __forceinline__ float siluf(float x){ return x / (1.0f + __expf(-x)); }
static __device__ __forceinline__ v16h cat8(half8 a, half8 b){
  return __builtin_shufflevector(a, b, 0,1,2,3,4,5,6,7,8,9,10,11,12,13,14,15);
}

#if defined(HAVE_TDM)
// TDM: load a 128-row x 32-col f16 tile (row stride = strideElems) into LDS.
// D# per CDNA5 ISA ch.8: group0 = {count/type/addrs}, group1 = {dims/strides}.
// This toolchain's builtin is the 6-arg form:
//   (uint32x4 g0, int32x8 g1, int32x4, int32x4, int32x8, i32 cpol)
static __device__ __forceinline__ void tdm_load_tile(uint32_t lds_off,
                                                     const _Float16* gptr,
                                                     int strideElems){
  uint64_t ga = (uint64_t)(uintptr_t)gptr;
  v4u g0;
  g0[0] = 1u;                                                  // count=1, user desc
  g0[1] = lds_off;                                             // lds_addr (bytes)
  g0[2] = (uint32_t)ga;                                        // global_addr[31:0]
  g0[3] = (uint32_t)((ga >> 32) & 0x01FFFFFFu) | (2u << 30);   // addr[56:32] | type=2
  v8i g1;
  g1[0] = (1 << 16);        // workgroup_mask=0, data_size=1 (2 bytes)
  g1[1] = (32 << 16);       // tensor_dim0[15:0]=32 (bits 63:48)
  g1[2] = (128 << 16);      // tensor_dim0[31:16]=0, tensor_dim1[15:0]=128
  g1[3] = (32 << 16);       // tensor_dim1[31:16]=0, tile_dim0=32
  g1[4] = 128;              // tile_dim1=128, tile_dim2=0
  g1[5] = strideElems;      // tensor_dim0_stride[31:0]
  g1[6] = 0;                // stride hi, tensor_dim1_stride lo
  g1[7] = 0;
  v4i z4 = {0,0,0,0};
  v8i z8 = {0,0,0,0,0,0,0,0};
  __builtin_amdgcn_tensor_load_to_lds(g0, g1, z4, z4, z8, 0);
}
#endif

// ---------------- elementwise helpers ----------------
__global__ void k_silu_vec(const float* __restrict__ s, float* __restrict__ d, int n){
  int i = blockIdx.x*blockDim.x + threadIdx.x;
  if (i < n) d[i] = siluf(s[i]);
}

// Weight swizzle: f32 [K x N] row-major -> f16 WMMA B-operand tiles.
__global__ void k_wswz(const float* __restrict__ W, _Float16* __restrict__ out, int K, int N){
  size_t total = (size_t)K*N;
  size_t st = (size_t)gridDim.x*blockDim.x;
  int ntk = K >> 5;
  for (size_t i = (size_t)blockIdx.x*blockDim.x + threadIdx.x; i < total; i += st){
    size_t tile = i >> 9;
    int rem  = (int)(i & 511);
    int lane = rem >> 4, j = rem & 15;
    int n0 = (int)(tile / ntk) << 4;
    int k0 = (int)(tile % ntk) << 5;
    int n  = n0 + (lane & 15);
    int k  = k0 + ((lane < 16) ? 0 : 8) + ((j < 8) ? j : (j + 8));
    out[i] = (_Float16)W[(size_t)k*N + n];
  }
}

__global__ void k_adaln(const float* __restrict__ sc, const float* __restrict__ W,
                        const float* __restrict__ bias, float* __restrict__ out){
  int j2 = blockIdx.x*blockDim.x + threadIdx.x;
  if (j2 >= 2*3*DM) return;
  int b = j2 / (3*DM), j = j2 % (3*DM);
  const float* scb = sc + b*DM;
  float acc = bias[j];
  for (int d = 0; d < DM; ++d) acc += scb[d] * W[d*(3*DM) + j];
  out[j2] = acc;
}

__global__ __launch_bounds__(256) void k_ln_mod(const float* __restrict__ X,
     const float* __restrict__ w, const float* __restrict__ bb,
     const float* __restrict__ ad, _Float16* __restrict__ out){
  int row = blockIdx.x;
  int b = row >> 10;
  const float* xr = X + (size_t)row*DM;
  __shared__ float s1[256], s2[256];
  float ls = 0.f, lq = 0.f;
  for (int d = threadIdx.x; d < DM; d += 256){ float v = xr[d]; ls += v; lq += v*v; }
  s1[threadIdx.x] = ls; s2[threadIdx.x] = lq; __syncthreads();
  for (int o = 128; o > 0; o >>= 1){
    if (threadIdx.x < o){ s1[threadIdx.x] += s1[threadIdx.x+o]; s2[threadIdx.x] += s2[threadIdx.x+o]; }
    __syncthreads();
  }
  float mean = s1[0] * (1.0f/DM);
  float var  = s2[0] * (1.0f/DM) - mean*mean;
  float rs = rsqrtf(var + 1e-5f);
  const float* shv = ad + b*3*DM;
  const float* sv  = shv + DM;
  _Float16* outr = out + (size_t)row*DM;
  for (int d = threadIdx.x; d < DM; d += 256){
    float v = (xr[d]-mean)*rs*w[d] + bb[d];
    v = v*(1.0f + sv[d]) + shv[d];
    outr[d] = (_Float16)v;
  }
}

// ---------------- WMMA GEMM, double-buffered DMA staging ----------------
// C[MxN] = A[MxK](f16 rm) @ B(pre-swizzled operand tiles), f32 out.
// Block 256 = 8 waves; block tile 128x64; wave tile 32x32 -> 4 WMMA/K-step.
// A tile staged ahead (k+1) into alternating LDS buffers by TDM (tensor_load_to_lds)
// or per-lane global_load_async_to_lds_b128; B fragments read straight from L2.
__global__ __launch_bounds__(256) void k_gemm(const _Float16* __restrict__ A,
    const _Float16* __restrict__ Bt, float* __restrict__ C, int M, int N, int K){
  __shared__ __align__(16) _Float16 As[2][128*32];
  const int tid = threadIdx.x, lane = tid & 31, wave = tid >> 5;
  const int wm = (wave & 3) << 5;
  const int wn = (wave >> 2) << 5;
  const int bm = blockIdx.y << 7, bn = blockIdx.x << 6;
  const int la = lane & 15;
  const int kA = (lane < 16) ? 0 : 8;
  const int ar = tid >> 1, ac = (tid & 1) << 4;
  const int ntk = K >> 5;
  v8f acc00 = {}, acc01 = {}, acc10 = {}, acc11 = {};
  const _Float16* Bp0 = Bt + ((size_t)((bn + wn)      >> 4) * ntk) * 512 + lane*16;
  const _Float16* Bp1 = Bt + ((size_t)((bn + wn + 16) >> 4) * ntk) * 512 + lane*16;
  const uint32_t ldsbase = (uint32_t)(uintptr_t)(lds_f16*)&As[0][0];

  // prologue: stage k-tile 0 into buffer 0
#if defined(HAVE_TDM)
  if (wave == 0) tdm_load_tile(ldsbase, A + (size_t)bm*K, K);
#else
  {
    const _Float16* gA = A + (size_t)(bm + ar)*K + ac;
    uint32_t dst = ldsbase + (uint32_t)(ar*32 + ac)*2;
    asm volatile("global_load_async_to_lds_b128 %0, %1, off" :: "v"(dst), "v"(gA) : "memory");
    asm volatile("global_load_async_to_lds_b128 %0, %1, off" :: "v"(dst + 16u), "v"(gA + 8) : "memory");
  }
#endif
  for (int kt = 0; kt < ntk; ++kt){
    const int cur = kt & 1;
    const uint32_t curoff = (uint32_t)cur * 8192u;
    // previous iteration's LDS fragment reads must be complete before the
    // other buffer is overwritten by the next stage (issued after barrier)
    asm volatile("s_wait_dscnt 0x0" ::: "memory");
#if defined(HAVE_TDM)
    if (wave == 0) __builtin_amdgcn_s_wait_tensorcnt(0);
#else
    asm volatile("s_wait_asynccnt 0x0" ::: "memory");
#endif
    __syncthreads();
    if (kt + 1 < ntk){
      const uint32_t nxtoff = 8192u - curoff;
#if defined(HAVE_TDM)
      if (wave == 0) tdm_load_tile(ldsbase + nxtoff, A + (size_t)bm*K + ((kt+1) << 5), K);
#else
      const _Float16* gA = A + (size_t)(bm + ar)*K + ((kt+1) << 5) + ac;
      uint32_t dst = ldsbase + nxtoff + (uint32_t)(ar*32 + ac)*2;
      asm volatile("global_load_async_to_lds_b128 %0, %1, off" :: "v"(dst), "v"(gA) : "memory");
      asm volatile("global_load_async_to_lds_b128 %0, %1, off" :: "v"(dst + 16u), "v"(gA + 8) : "memory");
#endif
    }
    // B fragments straight from global (pre-swizzled operand layout, L2-resident)
    const _Float16* p0 = Bp0 + (size_t)kt*512;
    const _Float16* p1 = Bp1 + (size_t)kt*512;
    __builtin_prefetch(p0 + 512, 0, 1);
    __builtin_prefetch(p1 + 512, 0, 1);
    v16h bf0 = cat8(*(const half8*)p0, *(const half8*)(p0 + 8));
    v16h bf1 = cat8(*(const half8*)p1, *(const half8*)(p1 + 8));
    // A fragments from current LDS buffer
    const _Float16* asb = &As[cur][0];
    half8 a00 = *(const half8*)&asb[(wm + la)*32 + kA];
    half8 a01 = *(const half8*)&asb[(wm + la)*32 + kA + 16];
    v16h af0 = cat8(a00, a01);
    half8 a10 = *(const half8*)&asb[(wm + 16 + la)*32 + kA];
    half8 a11 = *(const half8*)&asb[(wm + 16 + la)*32 + kA + 16];
    v16h af1 = cat8(a10, a11);
    acc00 = __builtin_amdgcn_wmma_f32_16x16x32_f16(false, af0, false, bf0, (short)0, acc00, false, false);
    acc01 = __builtin_amdgcn_wmma_f32_16x16x32_f16(false, af0, false, bf1, (short)0, acc01, false, false);
    acc10 = __builtin_amdgcn_wmma_f32_16x16x32_f16(false, af1, false, bf0, (short)0, acc10, false, false);
    acc11 = __builtin_amdgcn_wmma_f32_16x16x32_f16(false, af1, false, bf1, (short)0, acc11, false, false);
  }
  const int crow = bm + wm + ((lane < 16) ? 0 : 8);
  const int ccol = bn + wn + la;
  #pragma unroll
  for (int v = 0; v < 8; ++v){
    C[(size_t)(crow + v)*N      + ccol]      = acc00[v];
    C[(size_t)(crow + v)*N      + ccol + 16] = acc01[v];
    C[(size_t)(crow + 16 + v)*N + ccol]      = acc10[v];
    C[(size_t)(crow + 16 + v)*N + ccol + 16] = acc11[v];
  }
}

// ---------------- depthwise causal conv (both directions) + silu -> f16 ----------------
__global__ void k_conv_silu(const float* __restrict__ xz, const float* __restrict__ cw,
                            const float* __restrict__ cb, _Float16* __restrict__ out){
  size_t total = (size_t)4*L_*DI;
  size_t st = (size_t)gridDim.x*blockDim.x;
  for (size_t i = (size_t)blockIdx.x*blockDim.x + threadIdx.x; i < total; i += st){
    int d = (int)(i % DI);
    size_t row = i / DI;
    int t   = (int)(row & (L_-1));
    int bb  = (int)((row >> 10) & 1);
    int dir = (int)(row >> 11);
    float acc = cb[d];
    #pragma unroll
    for (int k = 0; k < 4; ++k){
      int tt = dir ? (t + 3 - k) : (t - 3 + k);
      if (tt >= 0 && tt < L_)
        acc += cw[d*4 + k] * xz[((size_t)bb*L_ + tt)*(2*DI) + d];
    }
    out[i] = (_Float16)siluf(acc);
  }
}

__global__ void k_dtin(const float* __restrict__ xdbl, _Float16* __restrict__ out){
  int i = blockIdx.x*blockDim.x + threadIdx.x;
  if (i >= 4096*DTR) return;
  int r = i >> 6, c = i & 63;
  out[i] = (_Float16)xdbl[(size_t)r*XPN + c];
}

__global__ void k_softplus(float* __restrict__ dt, const float* __restrict__ bias){
  size_t total = (size_t)4096*DI;
  size_t st = (size_t)gridDim.x*blockDim.x;
  for (size_t i = (size_t)blockIdx.x*blockDim.x + threadIdx.x; i < total; i += st){
    int d = (int)(i % DI);
    float v = dt[i] + bias[d];
    dt[i] = (v > 20.0f) ? v : log1pf(__expf(v));
  }
}

// ---------------- selective scan: thread per channel, uniform B/C loads ----------------
__global__ __launch_bounds__(256) void k_scan(const float* __restrict__ dt,
    const _Float16* __restrict__ xc, const float* __restrict__ xdbl,
    const float* __restrict__ A_log, const float* __restrict__ Dp,
    float* __restrict__ ysum, int dir, int addmode){
  int d = blockIdx.x*256 + threadIdx.x;
  int b = blockIdx.y;
  float Av[DS], h[DS];
  #pragma unroll
  for (int n = 0; n < DS; ++n){ Av[n] = -__expf(A_log[(size_t)d*DS + n]); h[n] = 0.f; }
  float Dpd = Dp[d];
  for (int s = 0; s < L_; ++s){
    int t = dir ? (L_-1-s) : s;
    size_t row = ((size_t)(dir*2 + b) << 10) + t;
    const float* bc = xdbl + row*XPN + DTR;   // wave-uniform address -> scalar loads
    float dtv = dt[row*DI + d];
    float xv  = (float)xc[row*DI + d];
    float dtx = dtv * xv;
    float y = Dpd * xv;
    #pragma unroll
    for (int n = 0; n < DS; ++n){
      h[n] = h[n]*__expf(dtv*Av[n]) + dtx*bc[n];
      y += h[n]*bc[DS + n];
    }
    size_t orow = ((size_t)b << 10) + t;
    if (addmode) ysum[orow*DI + d] += y;
    else         ysum[orow*DI + d]  = y;
  }
}

__global__ void k_gate(const float* __restrict__ ysum, const float* __restrict__ xz,
                       _Float16* __restrict__ g){
  size_t total = (size_t)2048*DI;
  size_t st = (size_t)gridDim.x*blockDim.x;
  for (size_t i = (size_t)blockIdx.x*blockDim.x + threadIdx.x; i < total; i += st){
    size_t row = i / DI; int d = (int)(i % DI);
    float z = xz[row*(2*DI) + DI + d];
    g[i] = (_Float16)(ysum[i] * siluf(z));
  }
}

__global__ void k_resid(const float* __restrict__ base, const float* __restrict__ delta,
                        const float* __restrict__ ad, float* __restrict__ out){
  size_t total = (size_t)2048*DM;
  size_t st = (size_t)gridDim.x*blockDim.x;
  for (size_t i = (size_t)blockIdx.x*blockDim.x + threadIdx.x; i < total; i += st){
    size_t row = i / DM; int b = (int)(row >> 10); int d = (int)(i % DM);
    out[i] = base[i] + ad[(size_t)b*3*DM + 2*DM + d] * delta[i];
  }
}

__global__ void k_silumul(const float* __restrict__ u1, const float* __restrict__ u2,
                          _Float16* __restrict__ out){
  size_t total = (size_t)2048*HID;
  size_t st = (size_t)gridDim.x*blockDim.x;
  for (size_t i = (size_t)blockIdx.x*blockDim.x + threadIdx.x; i < total; i += st)
    out[i] = (_Float16)(siluf(u1[i]) * u2[i]);
}

// ---------------- host ----------------
extern "C" void kernel_launch(void* const* d_in, const int* in_sizes, int n_in,
                              void* d_out, int out_size, void* d_ws, size_t ws_size,
                              hipStream_t stream){
  const float* x     = (const float*)d_in[0];
  const float* cond  = (const float*)d_in[1];
  const float* n1w   = (const float*)d_in[2];
  const float* n1b   = (const float*)d_in[3];
  const float* n2w   = (const float*)d_in[4];
  const float* n2b   = (const float*)d_in[5];
  const float* ad1W  = (const float*)d_in[6];
  const float* ad1b  = (const float*)d_in[7];
  const float* ad2W  = (const float*)d_in[8];
  const float* ad2b  = (const float*)d_in[9];
  const float* inW   = (const float*)d_in[10];
  const float* convw = (const float*)d_in[11];
  const float* convb = (const float*)d_in[12];
  const float* xpW   = (const float*)d_in[13];
  const float* dtW   = (const float*)d_in[14];
  const float* dtb   = (const float*)d_in[15];
  const float* A_log = (const float*)d_in[16];
  const float* Dp    = (const float*)d_in[17];
  const float* outW  = (const float*)d_in[18];
  const float* w1    = (const float*)d_in[19];
  const float* w2    = (const float*)d_in[20];
  const float* w3    = (const float*)d_in[21];
  float* out = (float*)d_out;
  (void)in_sizes; (void)n_in; (void)out_size; (void)ws_size;

  uint8_t* ws = (uint8_t*)d_ws;
  size_t o = 0;
  auto alloc = [&](size_t bytes)->void*{
    void* p = ws + o; o = (o + bytes + 255) & ~(size_t)255; return p;
  };
  _Float16* inW16  = (_Float16*)alloc((size_t)DM*2*DI*2);
  _Float16* xpW16  = (_Float16*)alloc((size_t)DI*XPN*2);
  _Float16* dtW16  = (_Float16*)alloc((size_t)DTR*DI*2);
  _Float16* outW16 = (_Float16*)alloc((size_t)DI*DM*2);
  _Float16* w1_16  = (_Float16*)alloc((size_t)DM*HID*2);
  _Float16* w2_16  = (_Float16*)alloc((size_t)DM*HID*2);
  _Float16* w3_16  = (_Float16*)alloc((size_t)HID*DM*2);
  float* sc   = (float*)alloc((size_t)2*DM*4);
  float* ad1  = (float*)alloc((size_t)2*3*DM*4);
  float* ad2  = (float*)alloc((size_t)2*3*DM*4);
  _Float16* h16    = (_Float16*)alloc((size_t)2048*DM*2);
  float* xz        = (float*)alloc((size_t)2048*2*DI*4);
  float* x1        = (float*)alloc((size_t)2048*DM*4);
  _Float16* xc16   = (_Float16*)alloc((size_t)4096*DI*2);
  float* xdbl      = (float*)alloc((size_t)4096*XPN*4);
  _Float16* dtin16 = (_Float16*)alloc((size_t)4096*DTR*2);
  float* dtbuf     = (float*)alloc((size_t)4096*DI*4);
  float* ysum      = (float*)alloc((size_t)2048*DI*4);
  _Float16* g16    = (_Float16*)alloc((size_t)2048*DI*2);
  float* mout      = (float*)alloc((size_t)2048*DM*4);
  float* u1        = (float*)alloc((size_t)2048*HID*4);
  float* u2        = (float*)alloc((size_t)2048*HID*4);
  _Float16* mlph16 = (_Float16*)alloc((size_t)2048*HID*2);
  float* mlp       = (float*)alloc((size_t)2048*DM*4);

  auto cg = [](size_t n, int b){ return (int)((n + (size_t)b - 1)/(size_t)b); };

  // 1) weight conversion + swizzle into WMMA operand tiles
  k_wswz<<<cg((size_t)DM*2*DI,256),256,0,stream>>>(inW,  inW16,  DM, 2*DI);
  k_wswz<<<cg((size_t)DI*XPN,256),256,0,stream>>>(xpW,  xpW16,  DI, XPN);
  k_wswz<<<cg((size_t)DTR*DI,256),256,0,stream>>>(dtW,  dtW16,  DTR, DI);
  k_wswz<<<cg((size_t)DI*DM,256),256,0,stream>>>(outW, outW16, DI, DM);
  k_wswz<<<cg((size_t)DM*HID,256),256,0,stream>>>(w1,  w1_16,  DM, HID);
  k_wswz<<<cg((size_t)DM*HID,256),256,0,stream>>>(w2,  w2_16,  DM, HID);
  k_wswz<<<cg((size_t)HID*DM,256),256,0,stream>>>(w3,  w3_16,  HID, DM);

  // 2) adaln
  k_silu_vec<<<cg(2*DM,256),256,0,stream>>>(cond, sc, 2*DM);
  k_adaln<<<cg(2*3*DM,256),256,0,stream>>>(sc, ad1W, ad1b, ad1);
  k_adaln<<<cg(2*3*DM,256),256,0,stream>>>(sc, ad2W, ad2b, ad2);

  // 3) LN1 + modulate -> h16
  k_ln_mod<<<2048,256,0,stream>>>(x, n1w, n1b, ad1, h16);

  // 4) xz = h @ in_proj  [2048 x 4096]
  k_gemm<<<dim3(4096/64, 2048/128),256,0,stream>>>(h16, inW16, xz, 2048, 2*DI, DM);

  // 5) depthwise conv + silu, both directions -> xc16 [4096 x 2048]
  k_conv_silu<<<8192,256,0,stream>>>(xz, convw, convb, xc16);

  // 6) xdbl = xconv @ x_proj  [4096 x 128]
  k_gemm<<<dim3(XPN/64, 4096/128),256,0,stream>>>(xc16, xpW16, xdbl, 4096, XPN, DI);

  // 7) dt = softplus(xdbl[:, :64] @ dt_proj + b)  [4096 x 2048]
  k_dtin<<<cg((size_t)4096*DTR,256),256,0,stream>>>(xdbl, dtin16);
  k_gemm<<<dim3(DI/64, 4096/128),256,0,stream>>>(dtin16, dtW16, dtbuf, 4096, DI, DTR);
  k_softplus<<<8192,256,0,stream>>>(dtbuf, dtb);

  // 8) selective scan, fwd then bwd (accumulate)
  k_scan<<<dim3(DI/256, B_),256,0,stream>>>(dtbuf, xc16, xdbl, A_log, Dp, ysum, 0, 0);
  k_scan<<<dim3(DI/256, B_),256,0,stream>>>(dtbuf, xc16, xdbl, A_log, Dp, ysum, 1, 1);

  // 9) gate with silu(z), out_proj, residual with g1
  k_gate<<<4096,256,0,stream>>>(ysum, xz, g16);
  k_gemm<<<dim3(DM/64, 2048/128),256,0,stream>>>(g16, outW16, mout, 2048, DM, DI);
  k_resid<<<2048,256,0,stream>>>(x, mout, ad1, x1);

  // 10) LN2 + modulate -> h16 (reuse)
  k_ln_mod<<<2048,256,0,stream>>>(x1, n2w, n2b, ad2, h16);

  // 11) MLP: silu(h@w1) * (h@w2) @ w3, residual with g2 -> out
  k_gemm<<<dim3(HID/64, 2048/128),256,0,stream>>>(h16, w1_16, u1, 2048, HID, DM);
  k_gemm<<<dim3(HID/64, 2048/128),256,0,stream>>>(h16, w2_16, u2, 2048, HID, DM);
  k_silumul<<<4096,256,0,stream>>>(u1, u2, mlph16);
  k_gemm<<<dim3(DM/64, 2048/128),256,0,stream>>>(mlph16, w3_16, mlp, 2048, DM, HID);
  k_resid<<<2048,256,0,stream>>>(x1, mlp, ad2, out);
}
